// Codebook_65034394796697
// MI455X (gfx1250) — matrix operand order repaired and checked by reading, not compiled, not run on hardware
//
#include <hip/hip_runtime.h>
#include <hip/hip_bf16.h>
#include <math.h>

typedef __attribute__((ext_vector_type(16))) _Float16 v16h;
typedef __attribute__((ext_vector_type(8)))  float    v8f;

#define B_DIM  8192
#define IN_DIMC 1024
#define K_DIM  2048
#define D_DIM  256
#define TAU_F  1.0f
#define BN_EPS_F 1e-5f

// ---------------- CDNA5 async global->LDS DMA (ASYNCcnt-tracked) ----------------
// global_load_async_to_lds_b128: per-lane 16B DMA, LDS byte address in a VGPR,
// 64-bit global address in a VGPR pair (GV mode, saddr=off).
__device__ __forceinline__ void asy_cp16(const _Float16* g, _Float16* l) {
  const unsigned lds = (unsigned)(unsigned long long)l;  // low 32b of flat = LDS offset
  asm volatile("global_load_async_to_lds_b128 %0, %1, off"
               :: "v"(lds), "v"(g) : "memory");
}
__device__ __forceinline__ void asy_wait0() {
  asm volatile("s_wait_asynccnt 0x0" ::: "memory");
}

// ---------------- WMMA fragment helpers (CDNA5 layouts, wave32) ----------------

// A-matrix 16x32 f16, tile stored row-major [M][ldk] in LDS.
__device__ __forceinline__ v16h frag_a_ld(const _Float16* s, int ldk) {
  const int lane = threadIdx.x & 31;
  const int m = lane & 15;
  const int hi = (lane >> 4) << 3;
  const _Float16* p = s + m * ldk;
  v16h a;
  #pragma unroll
  for (int pr = 0; pr < 8; ++pr) {
    const int kb = (pr < 4 ? (pr << 1) : 16 + ((pr - 4) << 1)) + hi;
    a[2 * pr]     = p[kb];
    a[2 * pr + 1] = p[kb + 1];
  }
  return a;
}

// B-matrix 32x16 f16 (KxN), tile stored n-major [N][ldk] (K contiguous) in LDS.
__device__ __forceinline__ v16h frag_b_ld(const _Float16* s, int ldk) {
  const int lane = threadIdx.x & 31;
  const int n = lane & 15;
  const int hi = (lane >> 4) << 3;
  const _Float16* p = s + n * ldk + hi;
  v16h b;
  #pragma unroll
  for (int j = 0; j < 16; ++j) b[j] = p[j < 8 ? j : 8 + j];
  return b;
}

__device__ __forceinline__ v8f wmma_f16(v16h a, v16h b, v8f c) {
  return __builtin_amdgcn_wmma_f32_16x16x32_f16(
      false, a, false, b, (short)0, c, false, false);
}

// ---------------- kernels ----------------

__global__ void k_zero(float* __restrict__ p, int n) {
  int i = blockIdx.x * blockDim.x + threadIdx.x;
  if (i < n) p[i] = 0.f;
}

// E prep: e2[k] = ||E_k||^2 ; Ef = f16(E) [K][D] ; Eft = f16(E^T) [D][K].
__global__ void __launch_bounds__(256) k_eprep(const float* __restrict__ E,
                                               _Float16* __restrict__ Ef,
                                               _Float16* __restrict__ Eft,
                                               float* __restrict__ e2) {
  const int t = threadIdx.x, wave = t >> 5, lane = t & 31;
  const int row = blockIdx.x * 8 + wave;
  const int d0 = lane * 8;
  const float* ep = E + (size_t)row * D_DIM + d0;
  _Float16* op = Ef + (size_t)row * D_DIM + d0;
  float acc = 0.f;
  #pragma unroll
  for (int i = 0; i < 8; ++i) {
    float v = ep[i];
    _Float16 hv = (_Float16)v;
    op[i] = hv;
    Eft[(size_t)(d0 + i) * K_DIM + row] = hv;  // transposed copy
    acc += v * v;
  }
  #pragma unroll
  for (int off = 16; off; off >>= 1) acc += __shfl_xor(acc, off);
  if (lane == 0) e2[row] = acc;
}

// GEMM1: h = x@W + b (128x64 tile/block) + BN batch-stat partials.
__global__ void __launch_bounds__(256) k_gemm1(const float* __restrict__ x,
                                               const float* __restrict__ W,
                                               const float* __restrict__ bias,
                                               float* __restrict__ h,
                                               float* __restrict__ colsum,
                                               float* __restrict__ colsq) {
  __shared__ __align__(16) _Float16 As[128 * 32];
  __shared__ __align__(16) _Float16 Bs[64 * 32];
  __shared__ float s_sum[64], s_sq[64], s_bias[64];
  const int t = threadIdx.x;
  const int bn0 = blockIdx.x * 64;
  const int m0  = blockIdx.y * 128;
  if (t < 64) { s_sum[t] = 0.f; s_sq[t] = 0.f; s_bias[t] = bias[bn0 + t]; }

  const int wave = t >> 5, lane = t & 31;
  const int wm = wave >> 1, wn = wave & 1;
  v8f c[2][2] = {};

  for (int k0 = 0; k0 < IN_DIMC; k0 += 32) {
    {  // stage A: fp32 -> f16 in-flight
      const int row = t >> 1, c0 = (t & 1) * 16;
      const float* gp = x + (size_t)(m0 + row) * IN_DIMC + k0 + c0;
      if (k0 + 32 < IN_DIMC) __builtin_prefetch(gp + 32, 0, 1);
      _Float16* sp = As + row * 32 + c0;
      #pragma unroll
      for (int i = 0; i < 16; i += 4) {
        float4 v = *(const float4*)(gp + i);
        sp[i + 0] = (_Float16)v.x; sp[i + 1] = (_Float16)v.y;
        sp[i + 2] = (_Float16)v.z; sp[i + 3] = (_Float16)v.w;
      }
    }
    {  // stage B: W fp32 -> f16, n-major
      const int kk = t >> 3, n0 = (t & 7) * 8;
      const float* gp = W + (size_t)(k0 + kk) * D_DIM + bn0 + n0;
      #pragma unroll
      for (int i = 0; i < 8; ++i) Bs[(n0 + i) * 32 + kk] = (_Float16)gp[i];
    }
    __syncthreads();
    v16h a0 = frag_a_ld(As + (wm * 32 + 0)  * 32, 32);
    v16h a1 = frag_a_ld(As + (wm * 32 + 16) * 32, 32);
    v16h b0 = frag_b_ld(Bs + (wn * 32 + 0)  * 32, 32);
    v16h b1 = frag_b_ld(Bs + (wn * 32 + 16) * 32, 32);
    c[0][0] = wmma_f16(a0, b0, c[0][0]);
    c[0][1] = wmma_f16(a0, b1, c[0][1]);
    c[1][0] = wmma_f16(a1, b0, c[1][0]);
    c[1][1] = wmma_f16(a1, b1, c[1][1]);
    __syncthreads();
  }

  const int nlane = lane & 15;
  const int hi8 = (lane >> 4) * 8;
  #pragma unroll
  for (int mf = 0; mf < 2; ++mf) {
    #pragma unroll
    for (int nf = 0; nf < 2; ++nf) {
      const int nl = wn * 32 + nf * 16 + nlane;
      const int ng = bn0 + nl;
      float sum = 0.f, sq = 0.f;
      #pragma unroll
      for (int r = 0; r < 8; ++r) {
        const int mg = m0 + wm * 32 + mf * 16 + hi8 + r;
        float v = c[mf][nf][r] + s_bias[nl];
        h[(size_t)mg * D_DIM + ng] = v;
        sum += v; sq += v * v;
      }
      atomicAdd(&s_sum[nl], sum);
      atomicAdd(&s_sq[nl], sq);
    }
  }
  __syncthreads();
  if (t < 64) {
    atomicAdd(&colsum[bn0 + t], s_sum[t]);
    atomicAdd(&colsq[bn0 + t],  s_sq[t]);
  }
}

__global__ void k_finalize(const float* __restrict__ colsum,
                           const float* __restrict__ colsq,
                           const float* __restrict__ gamma,
                           const float* __restrict__ beta,
                           float* __restrict__ scale,
                           float* __restrict__ shift) {
  const int t = threadIdx.x;  // 256 = D
  const float mu  = colsum[t] * (1.0f / B_DIM);
  const float var = colsq[t] * (1.0f / B_DIM) - mu * mu;
  const float sc  = gamma[t] * rsqrtf(var + BN_EPS_F);
  scale[t] = sc;
  shift[t] = beta[t] - mu * sc;
}

__global__ void __launch_bounds__(256) k_bnorm(const float* __restrict__ h,
                                               const float* __restrict__ scale,
                                               const float* __restrict__ shift,
                                               _Float16* __restrict__ hn,
                                               float* __restrict__ h2) {
  __shared__ float s_sc[256], s_sh[256];
  const int t = threadIdx.x;
  s_sc[t] = scale[t]; s_sh[t] = shift[t];
  __syncthreads();
  const int wave = t >> 5, lane = t & 31;
  const int row = blockIdx.x * 8 + wave;
  const int d0 = lane * 8;
  const float* hp = h + (size_t)row * D_DIM + d0;
  _Float16* op = hn + (size_t)row * D_DIM + d0;
  float acc = 0.f;
  #pragma unroll
  for (int i = 0; i < 8; ++i) {
    float v = hp[i] * s_sc[d0 + i] + s_sh[d0 + i];
    op[i] = (_Float16)v;
    acc += v * v;
  }
  #pragma unroll
  for (int off = 16; off; off >>= 1) acc += __shfl_xor(acc, off);
  if (lane == 0) h2[row] = acc;
}

// logits: out = 2*(hn@E^T) - h2 - e2. 128x128 tile; async double-buffered staging.
__global__ void __launch_bounds__(256) k_logits(const _Float16* __restrict__ hn,
                                                const _Float16* __restrict__ Ef,
                                                const float* __restrict__ h2,
                                                const float* __restrict__ e2,
                                                float* __restrict__ out_logits) {
  __shared__ __align__(16) _Float16 As[2][128 * 32];
  __shared__ __align__(16) _Float16 Bs[2][128 * 32];
  __shared__ float s_h2[128], s_e2[128];
  const int t = threadIdx.x;
  const int kc0 = blockIdx.x * 128;
  const int m0  = blockIdx.y * 128;
  if (t < 128) { s_h2[t] = h2[m0 + t]; s_e2[t] = e2[kc0 + t]; }
  const int wave = t >> 5, lane = t & 31;
  const int wm = wave >> 1, wn = wave & 1;
  const int row = t >> 1, c0 = (t & 1) * 16;   // staging coords (16 halfs/thread)
  const _Float16* gA = hn + (size_t)(m0 + row) * D_DIM + c0;
  const _Float16* gB = Ef + (size_t)(kc0 + row) * D_DIM + c0;
  v8f c[2][4] = {};

  // prologue: async DMA tile 0 into buffer 0
  asy_cp16(gA,     &As[0][row * 32 + c0]);
  asy_cp16(gA + 8, &As[0][row * 32 + c0 + 8]);
  asy_cp16(gB,     &Bs[0][row * 32 + c0]);
  asy_cp16(gB + 8, &Bs[0][row * 32 + c0 + 8]);

  for (int s = 0; s < D_DIM / 32; ++s) {
    const int cur = s & 1;
    asy_wait0();          // own async DMAs done
    __syncthreads();      // everyone's DMAs done; prev buffer free
    if (s + 1 < D_DIM / 32) {
      const int d0 = (s + 1) * 32;
      asy_cp16(gA + d0,     &As[cur ^ 1][row * 32 + c0]);
      asy_cp16(gA + d0 + 8, &As[cur ^ 1][row * 32 + c0 + 8]);
      asy_cp16(gB + d0,     &Bs[cur ^ 1][row * 32 + c0]);
      asy_cp16(gB + d0 + 8, &Bs[cur ^ 1][row * 32 + c0 + 8]);
    }
    v16h a0 = frag_a_ld(&As[cur][(wm * 32 + 0)  * 32], 32);
    v16h a1 = frag_a_ld(&As[cur][(wm * 32 + 16) * 32], 32);
    #pragma unroll
    for (int nf = 0; nf < 4; ++nf) {
      v16h b = frag_b_ld(&Bs[cur][(wn * 64 + nf * 16) * 32], 32);
      c[0][nf] = wmma_f16(a0, b, c[0][nf]);
      c[1][nf] = wmma_f16(a1, b, c[1][nf]);
    }
  }

  const int nlane = lane & 15, hi8 = (lane >> 4) * 8;
  #pragma unroll
  for (int mf = 0; mf < 2; ++mf)
    #pragma unroll
    for (int nf = 0; nf < 4; ++nf) {
      const int nl = wn * 64 + nf * 16 + nlane;
      #pragma unroll
      for (int r = 0; r < 8; ++r) {
        const int ml = wm * 32 + mf * 16 + hi8 + r;
        out_logits[(size_t)(m0 + ml) * K_DIM + kc0 + nl] =
            2.0f * c[mf][nf][r] - s_h2[ml] - s_e2[nl];
      }
    }
}

// fused dual softmax + z = assign@E. Async double-buffered E tiles from Eft.
__global__ void __launch_bounds__(256) k_softmax_z(const float* __restrict__ u,
                                                   const _Float16* __restrict__ Eft,
                                                   float* __restrict__ logits,
                                                   float* __restrict__ zout) {
  __shared__ __align__(16) _Float16 As[2][32 * 32];    // 4 KB
  __shared__ __align__(16) _Float16 Bs[2][256 * 32];   // 32 KB
  __shared__ float s_m1[32], s_i1[32], s_m2[32], s_i2[32];
  const int t = threadIdx.x, wave = t >> 5, lane = t & 31;
  const int r0 = blockIdx.x * 32;

  // ---- phase 1: online (max,sum) for softmax(logits) and softmax((logits+g)/tau)
  for (int rr = 0; rr < 4; ++rr) {
    const int rl = rr * 8 + wave;
    const size_t base = (size_t)(r0 + rl) * K_DIM;
    float m1 = -INFINITY, s1 = 0.f, m2 = -INFINITY, s2 = 0.f;
    for (int j = 0; j < K_DIM / 32; ++j) {
      const int col = lane + j * 32;
      const float l = logits[base + col];
      const float uu = u[base + col];
      const float g = -__logf(-__logf(uu + 1e-10f) + 1e-10f);
      if (l > m1) { s1 = s1 * __expf(m1 - l) + 1.f; m1 = l; }
      else        { s1 += __expf(l - m1); }
      const float v2 = (l + g) * (1.0f / TAU_F);
      if (v2 > m2) { s2 = s2 * __expf(m2 - v2) + 1.f; m2 = v2; }
      else         { s2 += __expf(v2 - m2); }
    }
    #pragma unroll
    for (int off = 16; off; off >>= 1) {
      float om = __shfl_xor(m1, off), os = __shfl_xor(s1, off);
      float nm = fmaxf(m1, om);
      s1 = s1 * __expf(m1 - nm) + os * __expf(om - nm); m1 = nm;
      om = __shfl_xor(m2, off); os = __shfl_xor(s2, off);
      nm = fmaxf(m2, om);
      s2 = s2 * __expf(m2 - nm) + os * __expf(om - nm); m2 = nm;
    }
    if (lane == 0) { s_m1[rl] = m1; s_i1[rl] = 1.0f / s1;
                     s_m2[rl] = m2; s_i2[rl] = 1.0f / s2; }
  }
  __syncthreads();

  // ---- phase 2: probs in place + f16 gumbel assignments + WMMA vs E tiles
  const int wmf = wave & 1;
  const int nb  = (wave >> 1) * 64;
  const _Float16* gE = Eft + (size_t)t * K_DIM;  // thread t owns Eft row d=t
  v8f c[4] = {};

  // prologue: async DMA E tile 0 (Bs is n-major [256][32], contiguous from Eft)
  {
    _Float16* lp = &Bs[0][t * 32];
    asy_cp16(gE,      lp);      asy_cp16(gE + 8,  lp + 8);
    asy_cp16(gE + 16, lp + 16); asy_cp16(gE + 24, lp + 24);
  }

  for (int s = 0; s < K_DIM / 32; ++s) {
    const int cur = s & 1;
    const int k0 = s * 32;
    {  // compute As tile: probs written back in place, assignments -> LDS (packed)
      const int row = t >> 3, c0 = (t & 7) * 4;
      const size_t idx = (size_t)(r0 + row) * K_DIM + k0 + c0;
      float4 l4 = *(const float4*)(logits + idx);
      float4 u4 = *(const float4*)(u + idx);
      const float m1 = s_m1[row], i1 = s_i1[row], m2 = s_m2[row];
      float lv[4] = {l4.x, l4.y, l4.z, l4.w};
      float uv[4] = {u4.x, u4.y, u4.z, u4.w};
      float4 p4; float* pp = (float*)&p4;
      union { _Float16 hh[4]; uint2 u2; } pk;
      #pragma unroll
      for (int i = 0; i < 4; ++i) {
        pp[i] = __expf(lv[i] - m1) * i1;
        const float g = -__logf(-__logf(uv[i] + 1e-10f) + 1e-10f);
        pk.hh[i] = (_Float16)__expf((lv[i] + g) * (1.0f / TAU_F) - m2);
      }
      *(uint2*)(&As[cur][row * 32 + c0]) = pk.u2;
      *(float4*)(logits + idx) = p4;
    }
    asy_wait0();
    __syncthreads();
    if (s + 1 < K_DIM / 32) {
      _Float16* lp = &Bs[cur ^ 1][t * 32];
      const _Float16* gp = gE + k0 + 32;
      asy_cp16(gp,      lp);      asy_cp16(gp + 8,  lp + 8);
      asy_cp16(gp + 16, lp + 16); asy_cp16(gp + 24, lp + 24);
    }
    v16h a = frag_a_ld(&As[cur][(wmf * 16) * 32], 32);
    #pragma unroll
    for (int nf = 0; nf < 4; ++nf) {
      v16h b = frag_b_ld(&Bs[cur][(nb + nf * 16) * 32], 32);
      c[nf] = wmma_f16(a, b, c[nf]);
    }
  }

  // ---- phase 3: z scaled by 1/s2 per row
  const int nlane = lane & 15, hi8 = (lane >> 4) * 8;
  #pragma unroll
  for (int nf = 0; nf < 4; ++nf) {
    const int ng = nb + nf * 16 + nlane;
    #pragma unroll
    for (int r = 0; r < 8; ++r) {
      const int ml = wmf * 16 + hi8 + r;
      zout[(size_t)(r0 + ml) * D_DIM + ng] = c[nf][r] * s_i2[ml];
    }
  }
}

// ---------------- host launch ----------------

extern "C" void kernel_launch(void* const* d_in, const int* in_sizes, int n_in,
                              void* d_out, int out_size, void* d_ws, size_t ws_size,
                              hipStream_t stream) {
  (void)in_sizes; (void)n_in; (void)out_size; (void)ws_size;
  const float* x     = (const float*)d_in[0];
  const float* u     = (const float*)d_in[1];
  const float* W     = (const float*)d_in[2];
  const float* bias  = (const float*)d_in[3];
  const float* gamma = (const float*)d_in[4];
  const float* beta  = (const float*)d_in[5];
  const float* E     = (const float*)d_in[6];

  float* out_probs = (float*)d_out;                      // [B, K]
  float* out_z     = out_probs + (size_t)B_DIM * K_DIM;  // [B, D]

  char* ws = (char*)d_ws;
  float*    h   = (float*)(ws);                                   // 8 MB
  _Float16* hn  = (_Float16*)(ws + (size_t)8  * 1024 * 1024);     // 4 MB
  _Float16* Ef  = (_Float16*)(ws + (size_t)12 * 1024 * 1024);     // 1 MB
  _Float16* Eft = (_Float16*)(ws + (size_t)13 * 1024 * 1024);     // 1 MB
  float*    h2  = (float*)(ws + (size_t)14 * 1024 * 1024);        // 32 KB
  float*    e2  = h2 + B_DIM;
  float* colsum = e2 + K_DIM;
  float* colsq  = colsum + D_DIM;
  float* scale  = colsq + D_DIM;
  float* shift  = scale + D_DIM;

  k_zero<<<dim3(2), dim3(256), 0, stream>>>(colsum, 2 * D_DIM);
  k_eprep<<<dim3(K_DIM / 8), dim3(256), 0, stream>>>(E, Ef, Eft, e2);
  k_gemm1<<<dim3(D_DIM / 64, B_DIM / 128), dim3(256), 0, stream>>>(
      x, W, bias, h, colsum, colsq);
  k_finalize<<<dim3(1), dim3(256), 0, stream>>>(colsum, colsq, gamma, beta, scale, shift);
  k_bnorm<<<dim3(B_DIM / 8), dim3(256), 0, stream>>>(h, scale, shift, hn, h2);
  k_logits<<<dim3(K_DIM / 128, B_DIM / 128), dim3(256), 0, stream>>>(
      hn, Ef, h2, e2, out_probs);
  k_softmax_z<<<dim3(B_DIM / 32), dim3(256), 0, stream>>>(u, Eft, out_probs, out_z);
}
